// WindowAttention_42863773614283
// MI455X (gfx1250) — matrix operand order repaired and checked by reading, not compiled, not run on hardware
//
#include <hip/hip_runtime.h>
#include <hip/hip_bf16.h>

typedef __bf16 bf16_t;
typedef bf16_t v16bf __attribute__((ext_vector_type(16)));
typedef float  v8f   __attribute__((ext_vector_type(8)));

#define WMMA_BF16 __builtin_amdgcn_wmma_f32_16x16x32_bf16

union BF16Tile { v16bf v; uint4 u[2]; };

__device__ __forceinline__ unsigned short f2bf(float f) {
  unsigned u = __builtin_bit_cast(unsigned, f);
  u += 0x7FFFu + ((u >> 16) & 1u);          // round-to-nearest-even
  return (unsigned short)(u >> 16);
}

#define B_ 8
#define C_ 64
#define N_ 4096
#define SCALE_ 0.3779644730092272f          // 1/sqrt(7)

// ---- CDNA5 async global->LDS path (ASYNCcnt), guarded so compile never breaks
#if defined(__has_builtin)
#if __has_builtin(__builtin_amdgcn_global_load_async_to_lds_b128)
#define HAVE_ASYNC_LDS 1
#endif
#if __has_builtin(__builtin_amdgcn_s_wait_asynccnt)
#define HAVE_WAIT_ASYNC 1
#endif
#endif

#if HAVE_ASYNC_LDS
// Builtin prototype (from hipcc diagnostic): params are pointers to
// int __attribute__((vector_size(16))) in global (AS1) / LDS (AS3).
typedef int v4i_t __attribute__((vector_size(16)));
typedef __attribute__((address_space(1))) v4i_t* gv4i_p;
typedef __attribute__((address_space(3))) v4i_t* lv4i_p;
#endif

__device__ __forceinline__ void copy_b128(const void* g, void* l) {
#if HAVE_ASYNC_LDS
  // global_load_async_to_lds_b128: per-lane 16B; LDS addr = low 32 bits of
  // generic shared pointer (aperture layout: ADDR[31:0] == LDS offset).
  __builtin_amdgcn_global_load_async_to_lds_b128(
      (gv4i_p)(unsigned long long)(uintptr_t)g,
      (lv4i_p)(unsigned int)(uintptr_t)l,
      0, 0);
#else
  *(uint4*)l = *(const uint4*)g;
#endif
}

__device__ __forceinline__ void drain_async() {
#if HAVE_ASYNC_LDS
#if HAVE_WAIT_ASYNC
  __builtin_amdgcn_s_wait_asynccnt(0);
#else
  asm volatile("s_wait_asynccnt 0x0" ::: "memory");
#endif
#endif
}

// ---------------------------------------------------------------------------
// Pass 1: fused 1x1-conv QKV projection, fp32 -> bf16
//   Qb, Kb : [B][N][C]  (token-major)
//   Vb     : [B][C][N]  (channel-major, so V^T tiles are contiguous)
// ---------------------------------------------------------------------------
__global__ __launch_bounds__(256)
void qkv_proj_kernel(const float* __restrict__ x,
                     const float* __restrict__ Wq, const float* __restrict__ bq,
                     const float* __restrict__ Wk, const float* __restrict__ bk,
                     const float* __restrict__ Wv, const float* __restrict__ bv,
                     unsigned short* __restrict__ Qb,
                     unsigned short* __restrict__ Kb,
                     unsigned short* __restrict__ Vb)
{
  const int t  = threadIdx.x;
  const int n  = blockIdx.x * 256 + t;     // pixel (coalesced over lanes)
  const int o0 = blockIdx.y * 4;           // 4 output channels per block
  const int b  = blockIdx.z;

  float aq[4], ak[4], av[4];
  #pragma unroll
  for (int i = 0; i < 4; ++i) { aq[i] = bq[o0+i]; ak[i] = bk[o0+i]; av[i] = bv[o0+i]; }

  const float* xp = x + (size_t)b * C_ * N_ + n;
  #pragma unroll 4
  for (int c = 0; c < C_; ++c) {
    float xv = xp[(size_t)c * N_];
    #pragma unroll
    for (int i = 0; i < 4; ++i) {
      aq[i] = fmaf(Wq[(o0+i)*C_ + c], xv, aq[i]);
      ak[i] = fmaf(Wk[(o0+i)*C_ + c], xv, ak[i]);
      av[i] = fmaf(Wv[(o0+i)*C_ + c], xv, av[i]);
    }
  }

  const size_t nm = ((size_t)b * N_ + n) * C_ + o0;   // token-major offset
  uint2 qpk, kpk;
  qpk.x = (unsigned)f2bf(aq[0]) | ((unsigned)f2bf(aq[1]) << 16);
  qpk.y = (unsigned)f2bf(aq[2]) | ((unsigned)f2bf(aq[3]) << 16);
  kpk.x = (unsigned)f2bf(ak[0]) | ((unsigned)f2bf(ak[1]) << 16);
  kpk.y = (unsigned)f2bf(ak[2]) | ((unsigned)f2bf(ak[3]) << 16);
  *(uint2*)(Qb + nm) = qpk;
  *(uint2*)(Kb + nm) = kpk;
  #pragma unroll
  for (int i = 0; i < 4; ++i)
    Vb[((size_t)b * C_ + o0 + i) * N_ + n] = f2bf(av[i]);
}

// ---------------------------------------------------------------------------
// Stage one 64-key K tile + V^T tile into LDS (8KB each), 128 threads.
// ---------------------------------------------------------------------------
__device__ __forceinline__ void stage_kv(int t,
                                         const unsigned short* __restrict__ Kb,
                                         const unsigned short* __restrict__ Vb,
                                         size_t kbase, size_t vbase, int k0,
                                         unsigned short* Kd, unsigned short* Vd)
{
  const uint4* ksrc = (const uint4*)(Kb + kbase + (size_t)k0 * C_);  // contiguous
  uint4* kdst = (uint4*)Kd;
  uint4* vdst = (uint4*)Vd;
  #pragma unroll
  for (int i = 0; i < 4; ++i) {
    int idx = t + i * 128;                 // 512 x 16B chunks
    copy_b128(ksrc + idx, kdst + idx);
  }
  #pragma unroll
  for (int i = 0; i < 4; ++i) {
    int idx = t + i * 128;
    int o = idx >> 3, c8 = idx & 7;        // 64 channel-rows x 128B
    copy_b128((const uint4*)(Vb + vbase + (size_t)o * N_ + k0) + c8, vdst + idx);
  }
}

// ---------------------------------------------------------------------------
// Pass 2: fused flash attention. 4 waves/block, 16 query rows per wave,
// double-buffered 64-key tiles through LDS. bf16 WMMA, fp32 accumulate.
// ---------------------------------------------------------------------------
__global__ __launch_bounds__(128)
void flash_attn_kernel(const unsigned short* __restrict__ Qb,
                       const unsigned short* __restrict__ Kb,
                       const unsigned short* __restrict__ Vb,
                       float* __restrict__ out)
{
  __shared__ alignas(16) unsigned short Klds[2][64 * 64];   // [buf][key][c]
  __shared__ alignas(16) unsigned short Vlds[2][64 * 64];   // [buf][o][key]
  __shared__ alignas(16) unsigned short Plds[4][16 * 64];   // per-wave P scratch

  const int t  = threadIdx.x;
  const int w  = t >> 5;         // wave id 0..3
  const int l  = t & 31;         // lane
  const int lr = l & 15;
  const int hi = l >> 4;         // half-wave select
  const int b  = blockIdx.y;
  const int q0 = blockIdx.x * 64 + w * 16;   // this wave's first query row

  // --- load Q A-operands once (two 16x32 c-slices, A-matrix VGPR layout) ---
  BF16Tile qa0, qa1;
  {
    const unsigned short* qp = Qb + ((size_t)b * N_ + q0 + lr) * C_;
    qa0.u[0] = *(const uint4*)(qp + hi*8);
    qa0.u[1] = *(const uint4*)(qp + hi*8 + 16);
    qa1.u[0] = *(const uint4*)(qp + 32 + hi*8);
    qa1.u[1] = *(const uint4*)(qp + 32 + hi*8 + 16);
  }

  v8f acc[4];
  #pragma unroll
  for (int i = 0; i < 4; ++i) acc[i] = (v8f){0,0,0,0,0,0,0,0};
  float m[8], ls[8];
  #pragma unroll
  for (int e = 0; e < 8; ++e) { m[e] = -1e30f; ls[e] = 0.0f; }

  const size_t kbase = (size_t)b * N_ * C_;   // Kb token-major
  const size_t vbase = (size_t)b * C_ * N_;   // Vb channel-major
  const int NT = N_ / 64;

  // prologue: stage tile 0 into buffer 0
  stage_kv(t, Kb, Vb, kbase, vbase, 0, Klds[0], Vlds[0]);
  drain_async();
  __syncthreads();

  for (int kt = 0; kt < NT; ++kt) {
    const int cur = kt & 1;

    // issue next tile's loads now; latency hides behind this tile's compute
    if (kt + 1 < NT)
      stage_kv(t, Kb, Vb, kbase, vbase, (kt + 1) * 64,
               Klds[cur ^ 1], Vlds[cur ^ 1]);

    const unsigned short* Kc = Klds[cur];
    const unsigned short* Vc = Vlds[cur];

    // --- S = Q * K^T  (16 x 64), 2 WMMAs per 16-key column tile ---
    v8f s[4];
    #pragma unroll
    for (int j = 0; j < 4; ++j) {
      const unsigned short* kp = Kc + (j*16 + lr) * 64 + hi*8;  // B layout
      BF16Tile kb0, kb1;
      kb0.u[0] = *(const uint4*)(kp);
      kb0.u[1] = *(const uint4*)(kp + 16);
      kb1.u[0] = *(const uint4*)(kp + 32);
      kb1.u[1] = *(const uint4*)(kp + 48);
      v8f z = {0,0,0,0,0,0,0,0};
      v8f t0 = WMMA_BF16(false, qa0.v, false, kb0.v, (short)0, z,  false, false);
      s[j]   = WMMA_BF16(false, qa1.v, false, kb1.v, (short)0, t0, false, false);
    }

    // --- online softmax; C/D layout row = e + 8*hi, col = lane&15 ---
    #pragma unroll
    for (int e = 0; e < 8; ++e) {
      float v0 = fmaxf(fmaxf(s[0][e], s[1][e]), fmaxf(s[2][e], s[3][e])) * SCALE_;
      #pragma unroll
      for (int mask = 1; mask < 16; mask <<= 1)
        v0 = fmaxf(v0, __shfl_xor(v0, mask, 32));
      float mnew = fmaxf(m[e], v0);
      float corr = __expf(m[e] - mnew);
      m[e] = mnew;

      float part = 0.0f;
      unsigned short* prow = &Plds[w][(e + 8*hi) * 64 + lr];
      #pragma unroll
      for (int j = 0; j < 4; ++j) {
        float p = __expf(fmaf(s[j][e], SCALE_, -mnew));
        part += p;
        prow[j*16] = f2bf(p);
      }
      #pragma unroll
      for (int mask = 1; mask < 16; mask <<= 1)
        part += __shfl_xor(part, mask, 32);
      ls[e] = ls[e] * corr + part;
      #pragma unroll
      for (int jo = 0; jo < 4; ++jo) acc[jo][e] *= corr;
    }

    // same-wave DS ordering for P scratch (stop compiler reordering too)
    asm volatile("s_wait_dscnt 0" ::: "memory");

    // reload P in A-matrix layout (two 16x32 key-slices)
    BF16Tile pa0, pa1;
    {
      const unsigned short* pp = &Plds[w][lr * 64 + hi*8];
      pa0.u[0] = *(const uint4*)(pp);
      pa0.u[1] = *(const uint4*)(pp + 16);
      pa1.u[0] = *(const uint4*)(pp + 32);
      pa1.u[1] = *(const uint4*)(pp + 48);
    }

    // --- O += P * V^T  (16 x 64) ---
    #pragma unroll
    for (int jo = 0; jo < 4; ++jo) {
      const unsigned short* vp = Vc + (jo*16 + lr) * 64 + hi*8;  // B layout
      BF16Tile vb0, vb1;
      vb0.u[0] = *(const uint4*)(vp);
      vb0.u[1] = *(const uint4*)(vp + 16);
      vb1.u[0] = *(const uint4*)(vp + 32);
      vb1.u[1] = *(const uint4*)(vp + 48);
      acc[jo] = WMMA_BF16(false, pa0.v, false, vb0.v, (short)0, acc[jo], false, false);
      acc[jo] = WMMA_BF16(false, pa1.v, false, vb1.v, (short)0, acc[jo], false, false);
    }

    // publish next tile (async drains + block barrier)
    drain_async();
    __syncthreads();
  }

  // --- epilogue: normalize, write [B][N][O] flat (== reference reshape) ---
  #pragma unroll
  for (int e = 0; e < 8; ++e) {
    float r = 1.0f / ls[e];
    int q = q0 + e + 8*hi;
    float* op = out + ((size_t)b * N_ + q) * C_ + lr;
    #pragma unroll
    for (int jo = 0; jo < 4; ++jo)
      op[jo*16] = acc[jo][e] * r;
  }
}

// ---------------------------------------------------------------------------
extern "C" void kernel_launch(void* const* d_in, const int* in_sizes, int n_in,
                              void* d_out, int out_size, void* d_ws, size_t ws_size,
                              hipStream_t stream)
{
  const float* x  = (const float*)d_in[0];
  const float* Wq = (const float*)d_in[1];
  const float* bq = (const float*)d_in[2];
  const float* Wk = (const float*)d_in[3];
  const float* bk = (const float*)d_in[4];
  const float* Wv = (const float*)d_in[5];
  const float* bv = (const float*)d_in[6];
  float* out = (float*)d_out;

  unsigned short* Qb = (unsigned short*)d_ws;                 // 4 MB
  unsigned short* Kb = Qb + (size_t)B_ * N_ * C_;             // 4 MB
  unsigned short* Vb = Kb + (size_t)B_ * N_ * C_;             // 4 MB

  dim3 pg(N_ / 256, C_ / 4, B_);
  qkv_proj_kernel<<<pg, 256, 0, stream>>>(x, Wq, bq, Wk, bk, Wv, bv, Qb, Kb, Vb);

  dim3 ag(N_ / 64, B_);
  flash_attn_kernel<<<ag, 128, 0, stream>>>(Qb, Kb, Vb, out);
}
